// ParallelRetention_76725295775755
// MI455X (gfx1250) — matrix operand order repaired and checked by reading, not compiled, not run on hardware
//
#include <hip/hip_runtime.h>
#include <hip/hip_bf16.h>
#include <math.h>

#define NROW 8192
#define DIM  512
#define EPSV 1e-5f
#define QSTR 520   // padded LDS row stride (elems) for Q tile: 260 dwords == 4 mod 64 banks
#define SSTR 136   // padded LDS row stride (elems) for S tile:  68 dwords == 4 mod 64 banks

typedef __attribute__((ext_vector_type(16))) __bf16    v16bf;
typedef __attribute__((ext_vector_type(8)))  float     v8f;
typedef __attribute__((ext_vector_type(4)))  unsigned int v4u;

// ---------------- helpers ----------------

__device__ __forceinline__ unsigned short f2bf(float f) {
    unsigned int u = __builtin_bit_cast(unsigned int, f);
    u += 0x7FFFu + ((u >> 16) & 1u);          // round-to-nearest-even
    return (unsigned short)(u >> 16);
}

__device__ __forceinline__ v8f vzero8() {
    v8f z;
#pragma unroll
    for (int i = 0; i < 8; ++i) z[i] = 0.0f;
    return z;
}

// A-fragment (16x32 bf16): lane<16 holds row=lane, K = {0..7, 16..23}; lane>=16 K = {8..15, 24..31}.
// p points at (row, kbase + lh*8); chunks are 8 elems at +0 and +16 elems.
__device__ __forceinline__ v16bf load_fragA(const unsigned short* p) {
    union { v4u u[2]; v16bf v; } f;
    f.u[0] = *(const v4u*)(p);
    f.u[1] = *(const v4u*)(p + 16);
    return f.v;
}

// B-fragment (32x16 bf16): lane holds column n=lane&15, K = {0..7, 16..23} for lane<16, {8..15, 24..31} for lane>=16.
__device__ __forceinline__ v16bf load_fragB(const unsigned short* p) {
    union { v4u u[2]; v16bf v; } f;
    f.u[0] = *(const v4u*)(p);
    f.u[1] = *(const v4u*)(p + 16);
    return f.v;
}

__device__ __forceinline__ v8f wmma_bf16(v16bf a, v16bf b, v8f c) {
    return __builtin_amdgcn_wmma_f32_16x16x32_bf16(false, a, false, b, (short)0, c, false, false);
}

// ---------------- pack kernels ----------------

__global__ void cvt_bf16_kernel(const float* __restrict__ x,
                                unsigned short* __restrict__ y, int n) {
    int i = blockIdx.x * blockDim.x + threadIdx.x;
    if (i < n) y[i] = f2bf(x[i]);
}

// wt[n*DIM + k] = bf16(w[k*DIM + n])
__global__ void transpose_bf16_kernel(const float* __restrict__ w,
                                      unsigned short* __restrict__ wt) {
    int i = blockIdx.x * blockDim.x + threadIdx.x;
    int n = i >> 9, k = i & (DIM - 1);
    wt[i] = f2bf(w[k * DIM + n]);
}

// ---------------- unified GEMM: Y = A(8192x512,bf16) @ W^T(512x512,bf16 as nxk) + bias ----------------
// MODE 0: store bf16 row-major   MODE 1: store bf16 transposed (n x NROW)
// MODE 2: gelu -> bf16 row-major MODE 3: fp32 + groupnorm -> Yf
template <int MODE>
__global__ __launch_bounds__(128, 1)
void gemm_bf16_kernel(const unsigned short* __restrict__ A,
                      const unsigned short* __restrict__ Wt,
                      const float* __restrict__ bias,
                      unsigned short* __restrict__ Yb,
                      float* __restrict__ Yf,
                      const float* __restrict__ gamma,
                      const float* __restrict__ beta) {
    __shared__ float lds[(MODE == 3) ? 16 * DIM : 16];
    const int i0   = blockIdx.x * 16;
    const int wave = threadIdx.x >> 5;
    const int lane = threadIdx.x & 31;
    const int ln = lane & 15, lh = lane >> 4;

    v8f acc[8];
#pragma unroll
    for (int t = 0; t < 8; ++t) acc[t] = vzero8();

    const unsigned short* arow = A + (size_t)(i0 + ln) * DIM + lh * 8;
#pragma unroll 4
    for (int ks = 0; ks < 16; ++ks) {
        v16bf af = load_fragA(arow + ks * 32);
#pragma unroll
        for (int t = 0; t < 8; ++t) {
            int n = wave * 128 + t * 16 + ln;
            v16bf bfr = load_fragB(Wt + (size_t)n * DIM + ks * 32 + lh * 8);
            acc[t] = wmma_bf16(af, bfr, acc[t]);
        }
    }

#pragma unroll
    for (int t = 0; t < 8; ++t) {
        float b = bias[wave * 128 + t * 16 + ln];
#pragma unroll
        for (int r = 0; r < 8; ++r) acc[t][r] += b;
    }

    if (MODE == 0 || MODE == 2) {
#pragma unroll
        for (int t = 0; t < 8; ++t) {
            int n = wave * 128 + t * 16 + ln;
#pragma unroll
            for (int r = 0; r < 8; ++r) {
                float v = acc[t][r];
                if (MODE == 2) v = 0.5f * v * (1.0f + erff(v * 0.7071067811865475f));
                Yb[(size_t)(i0 + lh * 8 + r) * DIM + n] = f2bf(v);
            }
        }
    } else if (MODE == 1) {
        // transposed store: rows contiguous per lane -> one b128 per tile
#pragma unroll
        for (int t = 0; t < 8; ++t) {
            int n = wave * 128 + t * 16 + ln;
            v4u pk;
#pragma unroll
            for (int q = 0; q < 4; ++q)
                pk[q] = (unsigned int)f2bf(acc[t][2 * q]) |
                        ((unsigned int)f2bf(acc[t][2 * q + 1]) << 16);
            *(v4u*)(Yb + (size_t)n * NROW + i0 + lh * 8) = pk;
        }
    } else { // MODE 3: groupnorm epilogue
#pragma unroll
        for (int t = 0; t < 8; ++t) {
            int n = wave * 128 + t * 16 + ln;
#pragma unroll
            for (int r = 0; r < 8; ++r) lds[(lh * 8 + r) * DIM + n] = acc[t][r];
        }
        __syncthreads();
        for (int it = threadIdx.x; it < 256; it += 128) { // 16 rows x 16 groups
            int row = it >> 4, g = it & 15;
            const float* p = lds + row * DIM + g * 32;
            float s = 0.f, ss = 0.f;
#pragma unroll
            for (int c = 0; c < 32; ++c) { float v = p[c]; s += v; ss += v * v; }
            float mu  = s * (1.0f / 32.0f);
            float var = ss * (1.0f / 32.0f) - mu * mu;
            float inv = rsqrtf(var + EPSV);
#pragma unroll
            for (int c = 0; c < 32; ++c) {
                int ch = g * 32 + c;
                Yf[(size_t)(i0 + row) * DIM + ch] = (p[c] - mu) * inv * gamma[ch] + beta[ch];
            }
        }
    }
}

// ---------------- fused retention: Xret = (D * (Q K^T)) @ V ----------------
// Block = 32 rows (2 row-tiles), 8 waves (256 threads). Q tile in bank-padded
// LDS. Per 128-col j-strip: phase A, wave w owns j-slab [j0+16w, +16) and
// computes BOTH row-tiles' S tiles off a single K fragment load (K read once
// per block per strip); D-scaled bf16 S goes to double-buffered LDS. One
// barrier. Phase B, wave = (row-group rg, col-wave cw): consumes the 16x128
// S slab of its row-group as A-fragments against V^T B-fragments.
__global__ __launch_bounds__(256, 1)
void retention_kernel(const unsigned short* __restrict__ Qb,
                      const unsigned short* __restrict__ Kb,
                      const unsigned short* __restrict__ Vt,
                      const float* __restrict__ Dm,
                      unsigned short* __restrict__ Xret) {
    __shared__ unsigned short Q_lds[32 * QSTR];     // ~33.3 KB
    __shared__ unsigned short S_lds[2][32 * SSTR];  // ~17.4 KB
    const int i0   = blockIdx.x * 32;
    const int wave = threadIdx.x >> 5;      // 0..7
    const int lane = threadIdx.x & 31;
    const int ln = lane & 15, lh = lane >> 4;
    const int rg = wave >> 2;               // phase-B row group (0..1)
    const int cw = wave & 3;                // phase-B col wave  (0..3)

    // cooperative fill of the 32x512 Q row-tile into padded LDS
    for (int t = threadIdx.x; t < 2048; t += 256) {
        int row = t >> 6, c = (t & 63) * 8;
        v4u d = *(const v4u*)(Qb + (size_t)(i0 + row) * DIM + c);
        *(v4u*)(&Q_lds[row * QSTR + c]) = d;
    }
    __syncthreads();

    v8f acc[8];
#pragma unroll
    for (int t = 0; t < 8; ++t) acc[t] = vzero8();

    const unsigned short* q0base = Q_lds + ln * QSTR + lh * 8;          // row-tile 0
    const unsigned short* q1base = Q_lds + (16 + ln) * QSTR + lh * 8;   // row-tile 1

    for (int j0 = 0; j0 < NROW; j0 += 128) {
        const int j1 = j0 + wave * 16;     // this wave's 16-col j-slab
        v8f s0 = vzero8(), s1 = vzero8();
        const unsigned short* kp = Kb + (size_t)(j1 + ln) * DIM + lh * 8;
#pragma unroll 2
        for (int ks = 0; ks < 16; ++ks) {
            v16bf kf = load_fragB(kp + ks * 32);   // shared by both row-tiles
            union { v4u u[2]; v16bf v; } qf0, qf1;
            const unsigned short* qp0 = q0base + ks * 32;
            const unsigned short* qp1 = q1base + ks * 32;
            qf0.u[0] = *(const v4u*)(qp0);
            qf0.u[1] = *(const v4u*)(qp0 + 16);
            qf1.u[0] = *(const v4u*)(qp1);
            qf1.u[1] = *(const v4u*)(qp1 + 16);
            s0 = wmma_bf16(qf0.v, kf, s0);
            s1 = wmma_bf16(qf1.v, kf, s1);
        }

        const int buf = (j0 >> 7) & 1;
        unsigned short* sl = &S_lds[buf][0];
        const int scol = wave * 16 + ln;   // strip column (0..127)
#pragma unroll
        for (int r = 0; r < 8; ++r) {
            int m = lh * 8 + r;            // row within a 16-row tile
            size_t dcol = j1 + ln;
            float v0 = s0[r] * Dm[(size_t)(i0 + m) * NROW + dcol];
            float v1 = s1[r] * Dm[(size_t)(i0 + 16 + m) * NROW + dcol];
            sl[m * SSTR + scol]        = f2bf(v0);
            sl[(16 + m) * SSTR + scol] = f2bf(v1);
        }
        __syncthreads();

        const unsigned short* srow = sl + (rg * 16 + ln) * SSTR + lh * 8;
#pragma unroll
        for (int kk = 0; kk < 4; ++kk) {
            const unsigned short* sp = srow + kk * 32;
            union { v4u u[2]; v16bf v; } sf;
            sf.u[0] = *(const v4u*)(sp);
            sf.u[1] = *(const v4u*)(sp + 16);
#pragma unroll
            for (int t = 0; t < 8; ++t) {
                int n = cw * 128 + t * 16 + ln;
                v16bf vb = load_fragB(Vt + (size_t)n * NROW + j0 + kk * 32 + lh * 8);
                acc[t] = wmma_bf16(sf.v, vb, acc[t]);
            }
        }
    }

#pragma unroll
    for (int t = 0; t < 8; ++t) {
        int n = cw * 128 + t * 16 + ln;
#pragma unroll
        for (int r = 0; r < 8; ++r)
            Xret[(size_t)(i0 + rg * 16 + lh * 8 + r) * DIM + n] = f2bf(acc[t][r]);
    }
}

// ---------------- launch ----------------

extern "C" void kernel_launch(void* const* d_in, const int* in_sizes, int n_in,
                              void* d_out, int out_size, void* d_ws, size_t ws_size,
                              hipStream_t stream) {
    const float* x  = (const float*)d_in[0];
    const float* Dm = (const float*)d_in[1];
    const float* Wq = (const float*)d_in[2];
    const float* bq = (const float*)d_in[3];
    const float* Wk = (const float*)d_in[4];
    const float* bk = (const float*)d_in[5];
    const float* Wv = (const float*)d_in[6];
    const float* bv = (const float*)d_in[7];
    const float* Wf = (const float*)d_in[8];
    const float* bf = (const float*)d_in[9];
    const float* Wp = (const float*)d_in[10];
    const float* bp = (const float*)d_in[11];
    const float* gamma = (const float*)d_in[12];
    const float* beta  = (const float*)d_in[13];
    float* out = (float*)d_out;

    unsigned short* ws  = (unsigned short*)d_ws;
    unsigned short* xb  = ws;                                  // 8192x512 bf16
    unsigned short* wqT = xb  + (size_t)NROW * DIM;            // 512x512 each
    unsigned short* wkT = wqT + (size_t)DIM * DIM;
    unsigned short* wvT = wkT + (size_t)DIM * DIM;
    unsigned short* wfT = wvT + (size_t)DIM * DIM;
    unsigned short* wpT = wfT + (size_t)DIM * DIM;
    unsigned short* Qb  = wpT + (size_t)DIM * DIM;             // 8192x512
    unsigned short* Kb  = Qb  + (size_t)NROW * DIM;
    unsigned short* Vt  = Kb  + (size_t)NROW * DIM;            // 512x8192 (transposed)
    unsigned short* Xr  = Vt  + (size_t)NROW * DIM;
    unsigned short* Hb  = Xr  + (size_t)NROW * DIM;

    cvt_bf16_kernel<<<(NROW * DIM) / 256, 256, 0, stream>>>(x, xb, NROW * DIM);
    transpose_bf16_kernel<<<(DIM * DIM) / 256, 256, 0, stream>>>(Wq, wqT);
    transpose_bf16_kernel<<<(DIM * DIM) / 256, 256, 0, stream>>>(Wk, wkT);
    transpose_bf16_kernel<<<(DIM * DIM) / 256, 256, 0, stream>>>(Wv, wvT);
    transpose_bf16_kernel<<<(DIM * DIM) / 256, 256, 0, stream>>>(Wf, wfT);
    transpose_bf16_kernel<<<(DIM * DIM) / 256, 256, 0, stream>>>(Wp, wpT);

    gemm_bf16_kernel<0><<<NROW / 16, 128, 0, stream>>>(xb, wqT, bq, Qb, nullptr, nullptr, nullptr);
    gemm_bf16_kernel<0><<<NROW / 16, 128, 0, stream>>>(xb, wkT, bk, Kb, nullptr, nullptr, nullptr);
    gemm_bf16_kernel<1><<<NROW / 16, 128, 0, stream>>>(xb, wvT, bv, Vt, nullptr, nullptr, nullptr);

    retention_kernel<<<NROW / 32, 256, 0, stream>>>(Qb, Kb, Vt, Dm, Xr);

    gemm_bf16_kernel<2><<<NROW / 16, 128, 0, stream>>>(Xr, wfT, bf, Hb, nullptr, nullptr, nullptr);
    gemm_bf16_kernel<3><<<NROW / 16, 128, 0, stream>>>(Hb, wpT, bp, nullptr, out, gamma, beta);
}
